// MobileRaftStereoModel_11845519802617
// MI455X (gfx1250) — matrix operand order repaired and checked by reading, not compiled, not run on hardware
//
#include <hip/hip_runtime.h>
#include <math.h>

// ---------------------------------------------------------------------------
// CDNA5 (gfx1250) wave32 WMMA types
// ---------------------------------------------------------------------------
typedef __attribute__((ext_vector_type(16))) _Float16 v16h;
typedef __attribute__((ext_vector_type(8)))  float    v8f;

#define ACT_NONE 0
#define ACT_RELU 1
#define ACT_SIG  2
#define ACT_TANH 3

__device__ __forceinline__ float apply_act(float v, int act)
{
    if      (act == ACT_RELU) v = fmaxf(v, 0.f);
    else if (act == ACT_SIG)  v = 1.f / (1.f + __expf(-v));
    else if (act == ACT_TANH) v = tanhf(v);
    return v;
}

// Fragment layouts per CDNA5 ISA 7.12.2 (wave32):
//   A (16x32 f16): lane m=l&15, hi=l>>4; half j -> K = (j<8 ? 8*hi+j : 8*hi+j+8)
//   B (32x16 f16): lane n=l&15, hi=l>>4; half j -> K as for A (this code uses K = 16*hi + j for both A and B)
//   C/D (16x16 f32): VGPR j -> M = j + 8*hi; lane -> N = l&15

// ---------------------------------------------------------------------------
// 1x1 conv == GEMM, templated on (input H*W, Cin) so channel-stride offsets
// fold into global_load immediates. 16(Cout) x 32(pixel) tile per wave.
// ---------------------------------------------------------------------------
template<int HWI, int CIN>
__global__ void k_conv1x1_wmma(const float* __restrict__ in, const float* __restrict__ wt,
                               const float* __restrict__ bias, float* __restrict__ out,
                               int Win, int Cout, int Hout, int Wout, int stride, int act)
{
    constexpr int Cin = CIN;
    constexpr size_t HWi = HWI;
    constexpr int CinFull = Cin & ~31;

    const int lane = threadIdx.x;
    const int col = lane & 15, hi = lane >> 4;
    const int nb = blockIdx.z;
    const int coutBase = blockIdx.y * 16;
    const int HWo = Hout * Wout;

    const int pix0 = blockIdx.x * 32 + col;
    const int pix1 = pix0 + 16;
    const bool pv0 = pix0 < HWo, pv1 = pix1 < HWo;
    int oh0 = pv0 ? pix0 / Wout : 0, ow0 = pv0 ? pix0 - oh0 * Wout : 0;
    int oh1 = pv1 ? pix1 / Wout : 0, ow1 = pv1 ? pix1 - oh1 * Wout : 0;

    const int mcout = coutBase + (lane & 15);
    const bool mvalid = mcout < Cout;
    // Invalid cout rows read row 0; products land only in epilogue-masked rows.
    const float* __restrict__ wRow = wt + (size_t)(mvalid ? mcout : 0) * Cin;
    __builtin_prefetch(wRow, 0, 1);          // -> global_prefetch_b8

    const float* __restrict__ inN = in + (size_t)nb * Cin * HWi;
    const float* __restrict__ bp0 = inN + (size_t)(oh0 * stride) * Win + ow0 * stride;
    const float* __restrict__ bp1 = inN + (size_t)(oh1 * stride) * Win + ow1 * stride;

    v8f acc0 = {}, acc1 = {};
    for (int cc = 0; cc < CinFull; cc += 32) {
        v16h a, b0, b1;
        const float* wp = wRow + cc + hi * 16;
        float4 wa = *(const float4*)(wp);
        float4 wb = *(const float4*)(wp + 4);
        float4 wc = *(const float4*)(wp + 8);
        float4 wd = *(const float4*)(wp + 12);
        a[0]  = (_Float16)wa.x; a[1]  = (_Float16)wa.y; a[2]  = (_Float16)wa.z; a[3]  = (_Float16)wa.w;
        a[4]  = (_Float16)wb.x; a[5]  = (_Float16)wb.y; a[6]  = (_Float16)wb.z; a[7]  = (_Float16)wb.w;
        a[8]  = (_Float16)wc.x; a[9]  = (_Float16)wc.y; a[10] = (_Float16)wc.z; a[11] = (_Float16)wc.w;
        a[12] = (_Float16)wd.x; a[13] = (_Float16)wd.y; a[14] = (_Float16)wd.z; a[15] = (_Float16)wd.w;
        const float* bq0 = bp0 + (size_t)(cc + hi * 16) * HWi;
        const float* bq1 = bp1 + (size_t)(cc + hi * 16) * HWi;
#pragma unroll
        for (int j = 0; j < 16; ++j) {
            float v0 = bq0[(size_t)j * HWi];     // j*HWi folds into load immediate
            float v1 = bq1[(size_t)j * HWi];
            b0[j] = (_Float16)(pv0 ? v0 : 0.f);
            b1[j] = (_Float16)(pv1 ? v1 : 0.f);
        }
        acc0 = __builtin_amdgcn_wmma_f32_16x16x32_f16(false, a, false, b0,
                                                      (short)0, acc0, false, false);
        acc1 = __builtin_amdgcn_wmma_f32_16x16x32_f16(false, a, false, b1,
                                                      (short)0, acc1, false, false);
    }
    if constexpr (CinFull < Cin) {   // compile-time tail (Cin = 16)
        constexpr int cc = CinFull;
        v16h a, b0, b1;
#pragma unroll
        for (int j = 0; j < 16; ++j) {
            int Ka = hi * 16 + j;
            int ca = cc + Ka;
            a[j] = (_Float16)wRow[ca < Cin ? ca : Cin - 1];
            int cb = cc + hi * 16 + j;
            bool ok = cb < Cin;
            int cbc = ok ? cb : Cin - 1;
            float v0 = bp0[(size_t)cbc * HWi];
            float v1 = bp1[(size_t)cbc * HWi];
            b0[j] = (_Float16)((pv0 && ok) ? v0 : 0.f);
            b1[j] = (_Float16)((pv1 && ok) ? v1 : 0.f);
        }
        acc0 = __builtin_amdgcn_wmma_f32_16x16x32_f16(false, a, false, b0,
                                                      (short)0, acc0, false, false);
        acc1 = __builtin_amdgcn_wmma_f32_16x16x32_f16(false, a, false, b1,
                                                      (short)0, acc1, false, false);
    }

#pragma unroll
    for (int j = 0; j < 8; ++j) {
        int co = coutBase + j + 8 * hi;
        if (co >= Cout) continue;
        float bco = bias[co];
        size_t ob = ((size_t)nb * Cout + co) * HWo;
        if (pv0) out[ob + pix0] = apply_act(acc0[j] + bco, act);
        if (pv1) out[ob + pix1] = apply_act(acc1[j] + bco, act);
    }
}

// ---------------------------------------------------------------------------
// 3x3 conv (pad=1), tap-major, templated on (input H*W, Cin). Inner loops are
// constant-stride loads with immediate offsets; spatial masks are post-load
// selects. 16 x 32 tile, 2 wmma per K-chunk.
// ---------------------------------------------------------------------------
template<int HWI, int CIN>
__global__ void k_conv3x3_wmma(const float* __restrict__ in, const float* __restrict__ wt,
                               const float* __restrict__ bias, float* __restrict__ out,
                               int Hin, int Win, int Cout, int Hout, int Wout,
                               int stride, int act)
{
    constexpr int Cin = CIN;
    constexpr size_t HWi = HWI;
    constexpr int KHW = 9;
    constexpr int CinFull = Cin & ~31;
    constexpr int pad = 1;

    const int lane = threadIdx.x;
    const int col = lane & 15, hi = lane >> 4;
    const int nb = blockIdx.z;
    const int coutBase = blockIdx.y * 16;
    const int HWo = Hout * Wout;

    const int pix0 = blockIdx.x * 32 + col;
    const int pix1 = pix0 + 16;
    const bool pv0 = pix0 < HWo, pv1 = pix1 < HWo;
    int oh0 = pv0 ? pix0 / Wout : 0, ow0 = pv0 ? pix0 - oh0 * Wout : 0;
    int oh1 = pv1 ? pix1 / Wout : 0, ow1 = pv1 ? pix1 - oh1 * Wout : 0;
    const int ihb0 = oh0 * stride - pad, iwb0 = ow0 * stride - pad;
    const int ihb1 = oh1 * stride - pad, iwb1 = ow1 * stride - pad;

    const int mcout = coutBase + (lane & 15);
    const bool mvalid = mcout < Cout;
    const float* __restrict__ wRow = wt + (size_t)(mvalid ? mcout : 0) * Cin * KHW;
    __builtin_prefetch(wRow, 0, 1);          // -> global_prefetch_b8

    const float* __restrict__ inN = in + (size_t)nb * Cin * HWi;

    v8f acc0 = {}, acc1 = {};
    int kh = 0, kw = 0;
    for (int tap = 0; tap < KHW; ++tap) {
        const int ih0 = ihb0 + kh, iw0 = iwb0 + kw;
        const int ih1 = ihb1 + kh, iw1 = iwb1 + kw;
        const bool ok0 = pv0 && ih0 >= 0 && ih0 < Hin && iw0 >= 0 && iw0 < Win;
        const bool ok1 = pv1 && ih1 >= 0 && ih1 < Hin && iw1 >= 0 && iw1 < Win;
        const float* __restrict__ bp0 = inN + (size_t)(ok0 ? ih0 : 0) * Win + (ok0 ? iw0 : 0);
        const float* __restrict__ bp1 = inN + (size_t)(ok1 ? ih1 : 0) * Win + (ok1 ? iw1 : 0);
        const float* __restrict__ wTap = wRow + tap;

        for (int cc = 0; cc < CinFull; cc += 32) {
            v16h a, b0, b1;
#pragma unroll
            for (int j = 0; j < 16; ++j) {
                int Ka = hi * 16 + j;
                a[j] = (_Float16)wTap[(size_t)(cc + Ka) * KHW];   // Ka*9 -> immediate
            }
#pragma unroll
            for (int j = 0; j < 16; ++j) {
                float v0 = bp0[(size_t)(cc + hi * 16 + j) * HWi]; // j*HWi -> immediate
                float v1 = bp1[(size_t)(cc + hi * 16 + j) * HWi];
                b0[j] = (_Float16)(ok0 ? v0 : 0.f);
                b1[j] = (_Float16)(ok1 ? v1 : 0.f);
            }
            acc0 = __builtin_amdgcn_wmma_f32_16x16x32_f16(false, a, false, b0,
                                                          (short)0, acc0, false, false);
            acc1 = __builtin_amdgcn_wmma_f32_16x16x32_f16(false, a, false, b1,
                                                          (short)0, acc1, false, false);
        }
        if constexpr (CinFull < Cin) {   // compile-time tail (Cin = 16 or 41)
            constexpr int cc = CinFull;
            v16h a, b0, b1;
#pragma unroll
            for (int j = 0; j < 16; ++j) {
                int Ka = hi * 16 + j;
                int ca = cc + Ka;
                a[j] = (_Float16)wTap[(size_t)(ca < Cin ? ca : Cin - 1) * KHW];
                int cb = cc + hi * 16 + j;
                bool okc = cb < Cin;
                int cbc = okc ? cb : Cin - 1;
                float v0 = bp0[(size_t)cbc * HWi];
                float v1 = bp1[(size_t)cbc * HWi];
                b0[j] = (_Float16)((ok0 && okc) ? v0 : 0.f);
                b1[j] = (_Float16)((ok1 && okc) ? v1 : 0.f);
            }
            acc0 = __builtin_amdgcn_wmma_f32_16x16x32_f16(false, a, false, b0,
                                                          (short)0, acc0, false, false);
            acc1 = __builtin_amdgcn_wmma_f32_16x16x32_f16(false, a, false, b1,
                                                          (short)0, acc1, false, false);
        }
        if (++kw == 3) { kw = 0; ++kh; }
    }

#pragma unroll
    for (int j = 0; j < 8; ++j) {
        int co = coutBase + j + 8 * hi;
        if (co >= Cout) continue;
        float bco = bias[co];
        size_t ob = ((size_t)nb * Cout + co) * HWo;
        if (pv0) out[ob + pix0] = apply_act(acc0[j] + bco, act);
        if (pv1) out[ob + pix1] = apply_act(acc1[j] + bco, act);
    }
}

// ---------------------------------------------------------------------------
// 7x7 stem conv, Cin=3, stride 1, pad 3, 256x320 input. Dense K-packing
// (147 -> 5 chunks). Fully constant decode: div-by-49/7 become magic mults.
// ---------------------------------------------------------------------------
__global__ void k_conv7_wmma(const float* __restrict__ in, const float* __restrict__ wt,
                             const float* __restrict__ bias, float* __restrict__ out,
                             int Cout, int act)
{
    constexpr int Cin = 3, KW = 7, KHW = 49, pad = 3;
    constexpr int Hin = 256, Win = 320;
    constexpr int Hout = 256, Wout = 320;
    constexpr int HWo = Hout * Wout;
    constexpr int Ktot = Cin * KHW;           // 147
    constexpr int chunks = (Ktot + 31) >> 5;  // 5

    const int lane = threadIdx.x;
    const int col = lane & 15, hi = lane >> 4;
    const int nb = blockIdx.z;
    const int coutBase = blockIdx.y * 16;

    const int pix0 = blockIdx.x * 32 + col;
    const int pix1 = pix0 + 16;
    const bool pv0 = pix0 < HWo, pv1 = pix1 < HWo;
    int oh0 = pix0 / Wout, ow0 = pix0 - oh0 * Wout;
    int oh1 = pix1 / Wout, ow1 = pix1 - oh1 * Wout;

    const int mcout = coutBase + (lane & 15);
    const bool mvalid = mcout < Cout;
    const float* __restrict__ wRow = wt + (size_t)(mvalid ? mcout : 0) * Ktot;
    __builtin_prefetch(wRow, 0, 1);
    const float* __restrict__ inN = in + (size_t)nb * Cin * Hin * Win;

    v8f acc0 = {}, acc1 = {};
    for (int kc = 0; kc < chunks; ++kc) {
        v16h a, b0, b1;
#pragma unroll
        for (int j = 0; j < 16; ++j) {
            int Ka = hi * 16 + j;
            int kg = (kc << 5) + Ka;
            a[j] = (_Float16)wRow[kg < Ktot ? kg : Ktot - 1];

            int kgb = (kc << 5) + hi * 16 + j;
            bool kok = kgb < Ktot;
            int kgbc = kok ? kgb : Ktot - 1;
            int cin = kgbc / KHW;                 // divide by constant
            int rem = kgbc - cin * KHW;
            int khh = rem / KW, kww = rem - khh * KW;
            const float* bc = inN + (size_t)cin * Hin * Win;
            int ih0 = oh0 - pad + khh, iw0 = ow0 - pad + kww;
            int ih1 = oh1 - pad + khh, iw1 = ow1 - pad + kww;
            bool ok0 = pv0 && kok && ih0 >= 0 && ih0 < Hin && iw0 >= 0 && iw0 < Win;
            bool ok1 = pv1 && kok && ih1 >= 0 && ih1 < Hin && iw1 >= 0 && iw1 < Win;
            int ih0c = ih0 < 0 ? 0 : (ih0 >= Hin ? Hin - 1 : ih0);
            int iw0c = iw0 < 0 ? 0 : (iw0 >= Win ? Win - 1 : iw0);
            int ih1c = ih1 < 0 ? 0 : (ih1 >= Hin ? Hin - 1 : ih1);
            int iw1c = iw1 < 0 ? 0 : (iw1 >= Win ? Win - 1 : iw1);
            float v0 = bc[(size_t)ih0c * Win + iw0c];
            float v1 = bc[(size_t)ih1c * Win + iw1c];
            b0[j] = (_Float16)(ok0 ? v0 : 0.f);
            b1[j] = (_Float16)(ok1 ? v1 : 0.f);
        }
        acc0 = __builtin_amdgcn_wmma_f32_16x16x32_f16(false, a, false, b0,
                                                      (short)0, acc0, false, false);
        acc1 = __builtin_amdgcn_wmma_f32_16x16x32_f16(false, a, false, b1,
                                                      (short)0, acc1, false, false);
    }

#pragma unroll
    for (int j = 0; j < 8; ++j) {
        int co = coutBase + j + 8 * hi;
        if (co >= Cout) continue;
        float bco = bias[co];
        size_t ob = ((size_t)nb * Cout + co) * HWo;
        if (pv0) out[ob + pix0] = apply_act(acc0[j] + bco, act);
        if (pv1) out[ob + pix1] = apply_act(acc1[j] + bco, act);
    }
}

// ---------------------------------------------------------------------------
// Correlation volume: cost[n,h,w,v] = sum_c lf[n,c,h,w]*rf[n,c,h,v] * scale
// Batched WcxWcxC GEMM per (n,h). All shapes constant -> immediate offsets.
// ---------------------------------------------------------------------------
template<int HC, int WC, int C>
__global__ void k_cost_wmma(const float* __restrict__ lf, const float* __restrict__ rf,
                            float* __restrict__ cost, float scale)
{
    constexpr size_t cs = (size_t)HC * WC;
    const int lane = threadIdx.x;
    const int col = lane & 15, hi = lane >> 4;
    const int nh = blockIdx.z;
    const int n = nh / HC, h = nh - n * HC;
    const int wBase = blockIdx.y * 16;
    const int vBase = blockIdx.x * 16;
    const float* __restrict__ lfp = lf + ((size_t)n * C * HC + h) * WC;
    const float* __restrict__ rfp = rf + ((size_t)n * C * HC + h) * WC;

    v8f acc = {};
    for (int kc = 0; kc < C; kc += 32) {
        v16h a, b;
#pragma unroll
        for (int j = 0; j < 16; ++j) {
            int Ka = hi * 16 + j;
            a[j] = (_Float16)lfp[(size_t)(kc + Ka) * cs + wBase + col];
            int Kb = hi * 16 + j;
            b[j] = (_Float16)rfp[(size_t)(kc + Kb) * cs + vBase + col];
        }
        acc = __builtin_amdgcn_wmma_f32_16x16x32_f16(false, a, false, b,
                                                     (short)0, acc, false, false);
    }
#pragma unroll
    for (int j = 0; j < 8; ++j) {
        int m = j + 8 * hi;   // w offset within tile
        cost[(((size_t)n * HC + h) * WC + wBase + m) * WC + vBase + col] = acc[j] * scale;
    }
}

// ---------------------------------------------------------------------------
// GroupNorm: pass 1 stats (mean, rstd) per (n, group); pass 2 fused affine+act
// ---------------------------------------------------------------------------
__global__ void k_gn_stats(const float* __restrict__ x, float* __restrict__ mv,
                           int C, int HW, int groups)
{
    int n = blockIdx.x / groups, g = blockIdx.x - n * groups;
    int cpg = C / groups;
    size_t base = ((size_t)n * C + (size_t)g * cpg) * HW;
    size_t cnt = (size_t)cpg * HW;
    float s = 0.f, ss = 0.f;
    for (size_t i = threadIdx.x; i < cnt; i += blockDim.x) {
        float v = x[base + i];
        s += v; ss += v * v;
    }
    __shared__ float sh[256], sh2[256];
    sh[threadIdx.x] = s; sh2[threadIdx.x] = ss;
    __syncthreads();
    for (int o = blockDim.x >> 1; o > 0; o >>= 1) {
        if ((int)threadIdx.x < o) { sh[threadIdx.x] += sh[threadIdx.x + o];
                                    sh2[threadIdx.x] += sh2[threadIdx.x + o]; }
        __syncthreads();
    }
    if (threadIdx.x == 0) {
        float mean = sh[0] / (float)cnt;
        float var  = sh2[0] / (float)cnt - mean * mean;
        mv[blockIdx.x * 2]     = mean;
        mv[blockIdx.x * 2 + 1] = rsqrtf(var + 1e-5f);
    }
}

__global__ void k_gn_apply(const float* __restrict__ x, float* __restrict__ y,
                           const float* __restrict__ mv, const float* __restrict__ gam,
                           const float* __restrict__ bet, int C, int HW, int groups, int act)
{
    size_t i = (size_t)blockIdx.x * blockDim.x + threadIdx.x;
    int c = (int)((i / HW) % C);
    int n = (int)(i / ((size_t)C * HW));
    int g = c / (C / groups);
    float mean = mv[(n * groups + g) * 2];
    float rstd = mv[(n * groups + g) * 2 + 1];
    float v = (x[i] - mean) * rstd * gam[c] + bet[c];
    if (act == ACT_RELU) v = fmaxf(v, 0.f);
    y[i] = v;
}

// ---------------------------------------------------------------------------
// Small glue kernels
// ---------------------------------------------------------------------------
__global__ void k_preprocess(const float* __restrict__ L, const float* __restrict__ R,
                             float* __restrict__ x, size_t half)
{
    size_t i = (size_t)blockIdx.x * blockDim.x + threadIdx.x;
    if (i >= 2 * half) return;
    float v = (i < half) ? L[i] : R[i - half];
    x[i] = 2.f * (v * (1.f / 255.f)) - 1.f;
}

__global__ void k_fill(float* __restrict__ p, float v, size_t n)
{
    size_t i = (size_t)blockIdx.x * blockDim.x + threadIdx.x;
    if (i < n) p[i] = v;
}

__global__ void k_add(const float* __restrict__ a, const float* __restrict__ b,
                      float* __restrict__ o, size_t n)
{
    size_t i = (size_t)blockIdx.x * blockDim.x + threadIdx.x;
    if (i < n) o[i] = a[i] + b[i];
}

__global__ void k_add_relu(const float* __restrict__ a, const float* __restrict__ b,
                           float* __restrict__ o, size_t n)
{
    size_t i = (size_t)blockIdx.x * blockDim.x + threadIdx.x;
    if (i < n) o[i] = fmaxf(a[i] + b[i], 0.f);
}

__global__ void k_mul(const float* __restrict__ a, const float* __restrict__ b,
                      float* __restrict__ o, size_t n)
{
    size_t i = (size_t)blockIdx.x * blockDim.x + threadIdx.x;
    if (i < n) o[i] = a[i] * b[i];
}

__global__ void k_gru_combine(const float* __restrict__ z, const float* __restrict__ h,
                              const float* __restrict__ q, float* __restrict__ o, size_t n)
{
    size_t i = (size_t)blockIdx.x * blockDim.x + threadIdx.x;
    if (i >= n) return;
    float zz = z[i];
    o[i] = (1.f - zz) * h[i] + zz * q[i];
}

__global__ void k_concat2(const float* __restrict__ a, const float* __restrict__ b,
                          float* __restrict__ o, int C1, int C2, int HW, size_t tot)
{
    size_t i = (size_t)blockIdx.x * blockDim.x + threadIdx.x;
    if (i >= tot) return;
    int Ct = C1 + C2;
    size_t n = i / ((size_t)Ct * HW);
    size_t rem = i - n * (size_t)Ct * HW;
    int c = (int)(rem / HW);
    size_t hw = rem - (size_t)c * HW;
    o[i] = (c < C1) ? a[((size_t)n * C1 + c) * HW + hw]
                    : b[((size_t)n * C2 + (c - C1)) * HW + hw];
}

__global__ void k_soft_argmin(const float* __restrict__ cv, float* __restrict__ flow,
                              int rows, int D)
{
    int i = blockIdx.x * blockDim.x + threadIdx.x;
    if (i >= rows) return;
    const float* row = cv + (size_t)i * D;
    float mx = row[0];
    for (int d = 1; d < D; ++d) mx = fmaxf(mx, row[d]);
    float s = 0.f, sd = 0.f;
    for (int d = 0; d < D; ++d) {
        float e = __expf(row[d] - mx);
        s += e; sd += e * (float)d;
    }
    flow[i] = sd / s;
}

__global__ void k_sample_cost(const float* __restrict__ cv, const float* __restrict__ flow,
                              float* __restrict__ out, int Hc, int Wc, int D, int rows)
{
    int i = blockIdx.x * blockDim.x + threadIdx.x;   // over N*Hc*Wc
    if (i >= rows) return;
    int n = i / (Hc * Wc);
    int hw = i - n * Hc * Wc;
    const float* row = cv + (size_t)i * D;
    float f0 = flow[i];
#pragma unroll
    for (int o = 0; o < 9; ++o) {
        float pos = f0 + (float)(o - 4);
        float x0f = floorf(pos);
        float fr = pos - x0f;
        int x0 = (int)x0f, x1 = x0 + 1;
        float v0 = (x0 >= 0 && x0 < D) ? row[x0] : 0.f;
        float v1 = (x1 >= 0 && x1 < D) ? row[x1] : 0.f;
        out[((size_t)(n * 9 + o) * Hc) * Wc + hw] = v0 * (1.f - fr) + v1 * fr;
    }
}

__global__ void k_avgpool_vol(const float* __restrict__ in, float* __restrict__ out,
                              int Ho, int Wo, int Do, size_t tot)
{
    size_t i = (size_t)blockIdx.x * blockDim.x + threadIdx.x;  // over N*Ho*Wo*Do
    if (i >= tot) return;
    int dd = (int)(i % Do); size_t t = i / Do;
    int wo = (int)(t % Wo); t /= Wo;
    int ho = (int)(t % Ho); int n = (int)(t / Ho);
    int Hi = 2 * Ho, Wi = 2 * Wo, Di = 2 * Do;
    const float* b = in + (((size_t)n * Hi + 2 * ho) * Wi + 2 * wo) * Di + 2 * dd;
    float s = 0.f;
#pragma unroll
    for (int dy = 0; dy < 2; ++dy)
#pragma unroll
        for (int dx = 0; dx < 2; ++dx)
#pragma unroll
            for (int dz = 0; dz < 2; ++dz)
                s += b[((size_t)dy * Wi + dx) * Di + dz];
    out[i] = s * 0.125f;
}

__global__ void k_resize(const float* __restrict__ in, float* __restrict__ out,
                         int Hi, int Wi, int Ho, int Wo, size_t tot)
{
    size_t i = (size_t)blockIdx.x * blockDim.x + threadIdx.x;  // over NC*Ho*Wo
    if (i >= tot) return;
    int ow = (int)(i % Wo); size_t t = i / Wo;
    int oh = (int)(t % Ho); int nc = (int)(t / Ho);
    float sy = (Ho > 1) ? (float)(Hi - 1) / (float)(Ho - 1) : (float)(Hi - 1);
    float sx = (Wo > 1) ? (float)(Wi - 1) / (float)(Wo - 1) : (float)(Wi - 1);
    float ys = oh * sy, xs = ow * sx;
    int y0 = (int)floorf(ys); int y1 = y0 + 1; if (y1 > Hi - 1) y1 = Hi - 1;
    int x0 = (int)floorf(xs); int x1 = x0 + 1; if (x1 > Wi - 1) x1 = Wi - 1;
    float wy = ys - (float)y0, wx = xs - (float)x0;
    const float* b = in + (size_t)nc * Hi * Wi;
    float v00 = b[(size_t)y0 * Wi + x0], v01 = b[(size_t)y0 * Wi + x1];
    float v10 = b[(size_t)y1 * Wi + x0], v11 = b[(size_t)y1 * Wi + x1];
    out[i] = v00 * (1.f - wy) * (1.f - wx) + v01 * (1.f - wy) * wx
           + v10 * wy * (1.f - wx) + v11 * wy * wx;
}

__global__ void k_warp(const float* __restrict__ img, const float* __restrict__ flow,
                       float* __restrict__ out, int C, int Hh, int Ww, size_t tot)
{
    size_t i = (size_t)blockIdx.x * blockDim.x + threadIdx.x;  // over N*C*Hh*Ww
    if (i >= tot) return;
    int ww = (int)(i % Ww); size_t t = i / Ww;
    int hh = (int)(t % Hh); t /= Hh;
    int c = (int)(t % C); int n = (int)(t / C);
    float fx = flow[((size_t)n * Hh + hh) * Ww + ww];
    float xp = ((float)ww + fx) * ((float)(Ww - 1) / (float)Ww);
    float yp = (float)hh * ((float)(Hh - 1) / (float)Hh);
    float x0f = floorf(xp), y0f = floorf(yp);
    float wx = xp - x0f, wy = yp - y0f;
    int x0 = (int)x0f, y0 = (int)y0f;
    const float* b = img + ((size_t)n * C + c) * Hh * Ww;
    auto tap = [&](int yi, int xi) -> float {
        if (xi < 0 || xi >= Ww || yi < 0 || yi >= Hh) return 0.f;
        return b[(size_t)yi * Ww + xi];
    };
    out[i] = tap(y0, x0)     * (1.f - wx) * (1.f - wy)
           + tap(y0, x0 + 1) * wx * (1.f - wy)
           + tap(y0 + 1, x0) * (1.f - wx) * wy
           + tap(y0 + 1, x0 + 1) * wx * wy;
}

// ---------------------------------------------------------------------------
// Host side
// ---------------------------------------------------------------------------
template<int V> struct IC { static constexpr int v = V; };

struct BlockP {
    const float *c1w,*c1b,*g1g,*g1b,*c2w,*c2b,*g2g,*g2b;
    const float *c3w,*c3b,*g3g,*g3b,*skw,*skb,*gsg,*gsb;
};
struct Params {
    const float *left, *right;
    const float *e0w0,*e0b0,*e0w1,*e0b1;
    BlockP blk[2];
    const float *wz,*bz,*wr,*br,*wq,*bq;
    const float *hw1,*hb1,*hw2,*hb2;
    const float *ew[4], *eb[4];
};

static void fill_sorted_params(Params& P, void* const* d_in, int base)
{
    auto f = [&](int i){ return (const float*)d_in[base + i]; };
    for (int bl = 0; bl < 2; ++bl) {
        BlockP& B = P.blk[bl]; int o = bl * 16;
        B.c1b = f(o+0);  B.c1w = f(o+1);  B.c2b = f(o+2);  B.c2w = f(o+3);
        B.c3b = f(o+4);  B.c3w = f(o+5);  B.g1b = f(o+6);  B.g1g = f(o+7);
        B.g2b = f(o+8);  B.g2g = f(o+9);  B.g3b = f(o+10); B.g3g = f(o+11);
        B.gsb = f(o+12); B.gsg = f(o+13); B.skb = f(o+14); B.skw = f(o+15);
    }
    P.e0b0 = f(32); P.e0b1 = f(33); P.e0w0 = f(34); P.e0w1 = f(35);
    P.eb[0] = f(36); P.eb[1] = f(37); P.eb[2] = f(38); P.eb[3] = f(39);
    P.ew[0] = f(40); P.ew[1] = f(41); P.ew[2] = f(42); P.ew[3] = f(43);
    P.bq = f(44); P.br = f(45); P.bz = f(46);
    P.wq = f(47); P.wr = f(48); P.wz = f(49);
    P.hb1 = f(50); P.hb2 = f(51); P.hw1 = f(52); P.hw2 = f(53);
}

static void fill_insertion_params(Params& P, void* const* d_in, int base)
{
    auto f = [&](int i){ return (const float*)d_in[base + i]; };
    P.e0w0 = f(0); P.e0b0 = f(1); P.e0w1 = f(2); P.e0b1 = f(3);
    for (int bl = 0; bl < 2; ++bl) {
        BlockP& B = P.blk[bl]; int o = 4 + bl * 16;
        B.c1w = f(o+0);  B.c1b = f(o+1);  B.g1g = f(o+2);  B.g1b = f(o+3);
        B.c2w = f(o+4);  B.c2b = f(o+5);  B.g2g = f(o+6);  B.g2b = f(o+7);
        B.c3w = f(o+8);  B.c3b = f(o+9);  B.g3g = f(o+10); B.g3b = f(o+11);
        B.skw = f(o+12); B.skb = f(o+13); B.gsg = f(o+14); B.gsb = f(o+15);
    }
    P.wz = f(36); P.bz = f(37); P.wr = f(38); P.br = f(39); P.wq = f(40); P.bq = f(41);
    P.hw1 = f(42); P.hb1 = f(43); P.hw2 = f(44); P.hb2 = f(45);
    P.ew[0] = f(46); P.eb[0] = f(47); P.ew[1] = f(48); P.eb[1] = f(49);
    P.ew[2] = f(50); P.eb[2] = f(51); P.ew[3] = f(52); P.eb[3] = f(53);
}

template<int HWI, int CIN>
static inline void conv1x1(hipStream_t s, const float* in, const float* w, const float* b,
                           float* out, int N, int Hin, int Win, int Cout, int stride, int act)
{
    int Hout = (Hin - 1) / stride + 1, Wout = (Win - 1) / stride + 1;
    dim3 grid((Hout * Wout + 31) / 32, (Cout + 15) / 16, N);
    k_conv1x1_wmma<HWI, CIN><<<grid, 32, 0, s>>>(in, w, b, out, Win, Cout,
                                                 Hout, Wout, stride, act);
}

template<int HWI, int CIN>
static inline void conv3x3(hipStream_t s, const float* in, const float* w, const float* b,
                           float* out, int N, int Hin, int Win, int Cout, int stride, int act)
{
    int Hout = (Hin - 1) / stride + 1, Wout = (Win - 1) / stride + 1;  // pad=1, K=3
    dim3 grid((Hout * Wout + 31) / 32, (Cout + 15) / 16, N);
    k_conv3x3_wmma<HWI, CIN><<<grid, 32, 0, s>>>(in, w, b, out, Hin, Win, Cout,
                                                 Hout, Wout, stride, act);
}

static inline void gnorm(hipStream_t s, float* x, const float* g, const float* b,
                         float* mv, int N, int C, int HW, int act)
{
    k_gn_stats<<<dim3(N * 8), 256, 0, s>>>(x, mv, C, HW, 8);
    size_t tot = (size_t)N * C * HW;   // always a multiple of 256 here
    k_gn_apply<<<dim3((unsigned)(tot / 256)), 256, 0, s>>>(x, x, mv, g, b, C, HW, 8, act);
}

#define GB(n) dim3((unsigned)(((size_t)(n) + 255) / 256)), dim3(256), 0, stream

extern "C" void kernel_launch(void* const* d_in, const int* in_sizes, int n_in,
                              void* d_out, int out_size, void* d_ws, size_t ws_size,
                              hipStream_t stream)
{
    const int H0 = 256, W0 = 320, H1 = 128, W1 = 160, H2 = 64, W2 = 80;
    const size_t LSZ = (size_t)2 * 3 * H0 * W0;   // 491520 floats per image input
    if (n_in < 56) return;

    // ---- resolve input-pointer convention at runtime ----
    Params P{};
    if (in_sizes[1] == (int)LSZ) {
        P.left = (const float*)d_in[0];
        P.right = (const float*)d_in[1];
        if (in_sizes[2] == 9408) fill_insertion_params(P, d_in, 2);   // insertion order
        else                     fill_sorted_params(P, d_in, 2);      // sorted params
    } else {
        P.left = (const float*)d_in[0];                               // full pytree sort
        P.right = (const float*)d_in[n_in - 1];
        fill_sorted_params(P, d_in, 1);
    }

    // ---- workspace bump allocator ----
    float* ws = (float*)d_ws;
    size_t off = 0;
    auto alloc = [&](size_t n) { float* p = ws + off; off += (n + 63) & ~(size_t)63; return p; };

    float* xin  = alloc((size_t)4 * 3 * H0 * W0);
    float* big0 = alloc((size_t)4 * 64 * H0 * W0);  // conv7 out; later 2x128 concat buffer
    float* pyr0 = alloc((size_t)4 * 64 * H0 * W0);
    float* pyr1 = alloc((size_t)4 * 64 * H1 * W1);
    float* pyr2 = alloc((size_t)4 * 64 * H2 * W2);
    float* tA   = alloc((size_t)2 * 64 * H0 * W0);  // block tmp / e ping
    float* tB   = alloc((size_t)2 * 64 * H0 * W0);  // block tmp / warp / e pong
    float* tC   = alloc((size_t)4 * 16 * H0 * W0);  // block y1 / skip
    float* cost0 = alloc((size_t)2 * 64 * 80 * 80);
    float* cv1   = alloc((size_t)2 * 32 * 40 * 40);
    float* cv2   = alloc((size_t)2 * 16 * 20 * 20);
    float* flow0 = alloc((size_t)2 * 64 * 80);
    float* flow1 = alloc((size_t)2 * 32 * 40);
    float* flow2 = alloc((size_t)2 * 16 * 20);
    float* cp0 = alloc((size_t)2 * 9 * 64 * 80);
    float* cp1 = alloc((size_t)2 * 9 * 32 * 40);
    float* cp2 = alloc((size_t)2 * 9 * 16 * 20);
    float* hstA = alloc((size_t)2 * 32 * 64 * 80);
    float* hstB = alloc((size_t)2 * 32 * 64 * 80);
    float* hx   = alloc((size_t)2 * 41 * 64 * 80);
    float* zb   = alloc((size_t)2 * 32 * 64 * 80);
    float* rb   = alloc((size_t)2 * 32 * 64 * 80);
    float* qb   = alloc((size_t)2 * 32 * 64 * 80);
    float* rh   = alloc((size_t)2 * 32 * 64 * 80);
    float* dfl  = alloc((size_t)2 * 64 * 80);
    float* edge = alloc((size_t)2 * H0 * W0);
    float* mv   = alloc(64);
    if (off * sizeof(float) > ws_size) return;      // workspace too small: bail

    // ================= Encoder =================
    k_preprocess<<<GB(2 * LSZ)>>>(P.left, P.right, xin, LSZ);
    {
        dim3 g((H0 * W0 + 31) / 32, 4, 4);
        k_conv7_wmma<<<g, 32, 0, stream>>>(xin, P.e0w0, P.e0b0, big0, 64, ACT_RELU);
    }
    conv1x1<81920, 64>(stream, big0, P.e0w1, P.e0b1, pyr0, 4, H0, W0, 64, 1, ACT_RELU);

    auto bottleneck = [&](auto tIn, auto tHalf, const float* x, float* out,
                          const BlockP& Bp, int Hin, int Win) {
        constexpr int HWIin = decltype(tIn)::v;
        constexpr int HWIh  = decltype(tHalf)::v;
        int Ho = Hin / 2, Wo = Win / 2;
        conv1x1<HWIin, 64>(stream, x, Bp.c1w, Bp.c1b, tC, 4, Hin, Win, 16, 1, ACT_NONE);
        gnorm(stream, tC, Bp.g1g, Bp.g1b, mv, 4, 16, Hin * Win, ACT_RELU);
        conv3x3<HWIin, 16>(stream, tC, Bp.c2w, Bp.c2b, tA, 4, Hin, Win, 16, 2, ACT_NONE);
        gnorm(stream, tA, Bp.g2g, Bp.g2b, mv, 4, 16, Ho * Wo, ACT_RELU);
        conv1x1<HWIh, 16>(stream, tA, Bp.c3w, Bp.c3b, tB, 4, Ho, Wo, 64, 1, ACT_NONE);
        gnorm(stream, tB, Bp.g3g, Bp.g3b, mv, 4, 64, Ho * Wo, ACT_NONE);
        conv1x1<HWIin, 64>(stream, x, Bp.skw, Bp.skb, tC, 4, Hin, Win, 64, 2, ACT_NONE);
        gnorm(stream, tC, Bp.gsg, Bp.gsb, mv, 4, 64, Ho * Wo, ACT_NONE);
        size_t n = (size_t)4 * 64 * Ho * Wo;
        k_add_relu<<<GB(n)>>>(tB, tC, out, n);
    };
    bottleneck(IC<81920>{}, IC<20480>{}, pyr0, pyr1, P.blk[0], H0, W0);
    bottleneck(IC<20480>{}, IC<5120>{},  pyr1, pyr2, P.blk[1], H1, W1);

    // ================= Correlation volume (WMMA batched GEMM) =================
    const float* lf2 = pyr2;
    const float* rf2 = pyr2 + (size_t)2 * 64 * H2 * W2;
    {
        dim3 g(W2 / 16, W2 / 16, 2 * H2);
        k_cost_wmma<64, 80, 64><<<g, 32, 0, stream>>>(lf2, rf2, cost0,
                                                      1.0f / sqrtf(64.0f));
    }

    // ================= Cost pyramid =================
    k_soft_argmin<<<GB(2 * 64 * 80)>>>(cost0, flow0, 2 * 64 * 80, 80);
    k_sample_cost<<<GB(2 * 64 * 80)>>>(cost0, flow0, cp0, 64, 80, 80, 2 * 64 * 80);
    k_avgpool_vol<<<GB(2 * 32 * 40 * 40)>>>(cost0, cv1, 32, 40, 40, (size_t)2 * 32 * 40 * 40);
    k_soft_argmin<<<GB(2 * 32 * 40)>>>(cv1, flow1, 2 * 32 * 40, 40);
    k_sample_cost<<<GB(2 * 32 * 40)>>>(cv1, flow1, cp1, 32, 40, 40, 2 * 32 * 40);
    k_avgpool_vol<<<GB(2 * 16 * 20 * 20)>>>(cv1, cv2, 16, 20, 20, (size_t)2 * 16 * 20 * 20);
    k_soft_argmin<<<GB(2 * 16 * 20)>>>(cv2, flow2, 2 * 16 * 20, 20);
    k_sample_cost<<<GB(2 * 16 * 20)>>>(cv2, flow2, cp2, 16, 20, 20, 2 * 16 * 20);

    // ================= ConvGRU over pyramid (coarse -> fine) =================
    float* hcur = hstA; float* hnext = hstB;
    k_fill<<<GB((size_t)2 * 32 * 16 * 20)>>>(hcur, 0.f, (size_t)2 * 32 * 16 * 20);
    auto gru_iter = [&](auto tag, int h, int w, float* cps) {
        constexpr int HWIv = decltype(tag)::v;
        int hw = h * w;
        size_t nhx = (size_t)2 * 41 * hw, nh32 = (size_t)2 * 32 * hw;
        k_concat2<<<GB(nhx)>>>(hcur, cps, hx, 32, 9, hw, nhx);
        conv3x3<HWIv, 41>(stream, hx, P.wz, P.bz, zb, 2, h, w, 32, 1, ACT_SIG);
        conv3x3<HWIv, 41>(stream, hx, P.wr, P.br, rb, 2, h, w, 32, 1, ACT_SIG);
        k_mul<<<GB(nh32)>>>(rb, hcur, rh, nh32);
        k_concat2<<<GB(nhx)>>>(rh, cps, hx, 32, 9, hw, nhx);
        conv3x3<HWIv, 41>(stream, hx, P.wq, P.bq, qb, 2, h, w, 32, 1, ACT_TANH);
        k_gru_combine<<<GB(nh32)>>>(zb, hcur, qb, hnext, nh32);
        { float* t = hcur; hcur = hnext; hnext = t; }
    };
    gru_iter(IC<320>{}, 16, 20, cp2);
    k_resize<<<GB((size_t)2 * 32 * 32 * 40)>>>(hcur, hnext, 16, 20, 32, 40,
                                               (size_t)2 * 32 * 32 * 40);
    { float* t = hcur; hcur = hnext; hnext = t; }
    gru_iter(IC<1280>{}, 32, 40, cp1);
    k_resize<<<GB((size_t)2 * 32 * 64 * 80)>>>(hcur, hnext, 32, 40, 64, 80,
                                               (size_t)2 * 32 * 64 * 80);
    { float* t = hcur; hcur = hnext; hnext = t; }
    gru_iter(IC<5120>{}, 64, 80, cp0);

    // ================= Flow head =================
    conv3x3<5120, 32>(stream, hcur, P.hw1, P.hb1, zb, 2, 64, 80, 32, 1, ACT_RELU);
    conv3x3<5120, 32>(stream, zb, P.hw2, P.hb2, dfl, 2, 64, 80, 1, 1, ACT_NONE);
    float* outq = (float*)d_out;                         // flow_pyr[0] (2,1,64,80)
    float* outh = outq + (size_t)2 * 64 * 80;            // flow_pyr[1] (2,1,128,160)
    float* outf = outh + (size_t)2 * 128 * 160;          // flow_pyr[2] (2,1,256,320)
    k_add<<<GB((size_t)2 * 64 * 80)>>>(flow0, dfl, outq, (size_t)2 * 64 * 80);

    // ================= Refinement =================
    auto refine = [&](auto tag, float* flow, const float* pyr, int h, int w) {
        constexpr int HWIv = decltype(tag)::v;
        size_t hw = (size_t)h * w;
        const float* lf = pyr;
        const float* rf = pyr + (size_t)2 * 64 * hw;
        size_t nw = (size_t)2 * 64 * hw, ne = (size_t)2 * 128 * hw;
        k_warp<<<GB(nw)>>>(rf, flow, tB, 64, h, w, nw);
        k_concat2<<<GB(ne)>>>(lf, tB, big0, 64, 64, (int)hw, ne);
        conv3x3<HWIv, 128>(stream, big0, P.ew[0], P.eb[0], tA, 2, h, w, 64, 1, ACT_RELU);
        conv3x3<HWIv, 64>(stream, tA, P.ew[1], P.eb[1], tB, 2, h, w, 64, 1, ACT_RELU);
        conv3x3<HWIv, 64>(stream, tB, P.ew[2], P.eb[2], tA, 2, h, w, 64, 1, ACT_RELU);
        conv3x3<HWIv, 64>(stream, tA, P.ew[3], P.eb[3], edge, 2, h, w, 1, 1, ACT_NONE);
        k_add<<<GB(2 * hw)>>>(flow, edge, flow, 2 * hw);
    };
    k_resize<<<GB((size_t)2 * H1 * W1)>>>(outq, outh, 64, 80, H1, W1, (size_t)2 * H1 * W1);
    refine(IC<20480>{}, outh, pyr1, H1, W1);
    k_resize<<<GB((size_t)2 * H0 * W0)>>>(outh, outf, H1, W1, H0, W0, (size_t)2 * H0 * W0);
    refine(IC<81920>{}, outf, pyr0, H0, W0);

    (void)in_sizes; (void)out_size;
}
